// TensorNetwork_6476810682407
// MI455X (gfx1250) — matrix operand order repaired and checked by reading, not compiled, not run on hardware
//
#include <hip/hip_runtime.h>
#include <math.h>

// ---------------------------------------------------------------------------
// TensorNet forward for MI455X (gfx1250, wave32, WMMA).
// Bandwidth-bound: node tensors (23 MB) are L2-resident (192 MB L2); dense
// GEMMs use v_wmma_f32_16x16x32_f16 with 16x64 per-wave strips (A-frag reuse),
// LDS weight panels stored pre-swizzled so B-frags load as ds_load_b128 pairs,
// A-frags load as float4 (global_load_b128). Segment sums = f32 atomics.
// Workspace requirement: ~83.4M floats (~334 MB) of d_ws.
// ---------------------------------------------------------------------------

#define N_NODES 10000
#define N_EDGES 160000
#define HC      64
#define RB      32
#define H3      192
#define CUT_R   5.0f
#define PI_F    3.14159265358979323846f

typedef __attribute__((ext_vector_type(16))) _Float16 v16h;
typedef __attribute__((ext_vector_type(8)))  float    v8f;

// A-matrix (16x32 f16) per-lane half index -> K  (order used by this code, not the ISA table's):
//   idx 0..15 -> k = 16*h0 + idx       (contiguous run of 16, matches B)
// B-matrix (32x16 f16): lane (h0,lr) holds col lr+16*ct, k = 16*h0 + idx.
// Swizzled LDS layout keys each lane's 16 halfs contiguously.

__device__ __forceinline__ v16h load_afrag_f32(const float* __restrict__ arow,
                                               int kc, int h0) {
  const float4 a0 = *(const float4*)(arow + kc + 16 * h0);
  const float4 a1 = *(const float4*)(arow + kc + 16 * h0 + 4);
  const float4 a2 = *(const float4*)(arow + kc + 16 * h0 + 8);
  const float4 a3 = *(const float4*)(arow + kc + 16 * h0 + 12);
  v16h af;
  af[0] = (_Float16)a0.x;  af[1] = (_Float16)a0.y;
  af[2] = (_Float16)a0.z;  af[3] = (_Float16)a0.w;
  af[4] = (_Float16)a1.x;  af[5] = (_Float16)a1.y;
  af[6] = (_Float16)a1.z;  af[7] = (_Float16)a1.w;
  af[8] = (_Float16)a2.x;  af[9] = (_Float16)a2.y;
  af[10] = (_Float16)a2.z; af[11] = (_Float16)a2.w;
  af[12] = (_Float16)a3.x; af[13] = (_Float16)a3.y;
  af[14] = (_Float16)a3.z; af[15] = (_Float16)a3.w;
  return af;
}

// ---------------------------------------------------------------------------
// Generic GEMM: out[M x Nc] = act(in[M x K] @ W[K x Nc] + bias) * rowscale?
// act: 0 = identity, 1 = tanh.  K multiple of 32 (<=192), Nc multiple of 64.
// Block = 8 waves sharing one 64-column group; B panel staged in LDS (f16),
// pre-swizzled: element (k,c) -> [((k/32)*4 + c/16)*32 + (k%32/16)*16 + c%16][k%16].
// Each wave computes a 16x64 strip: one A-frag feeds 4 WMMAs per K-chunk.
// ---------------------------------------------------------------------------
__global__ void tn_gemm(const float* __restrict__ A, const float* __restrict__ W,
                        const float* __restrict__ bias, const float* __restrict__ rowscale,
                        float* __restrict__ out, int M, int K, int Nc, int act,
                        int ngroups /* = Nc/64 */) {
  __shared__ alignas(32) _Float16 Bsw[H3 * 64];  // up to 192x64 f16 = 24 KB

  const int mblk = blockIdx.x / ngroups;
  const int ng = blockIdx.x - mblk * ngroups;
  const int tid = threadIdx.x;

  // cooperative, coalesced global read; swizzled LDS write
  for (int p = tid; p < K * 64; p += 256) {
    const int k = p >> 6, c = p & 63;
    const int chunk = k >> 5, kr = k & 31;
    const int slot = ((chunk << 2) + (c >> 4)) * 32 + ((kr >> 4) << 4) + (c & 15);
    Bsw[slot * 16 + (kr & 15)] = (_Float16)W[(size_t)k * Nc + ng * 64 + c];
  }
  __syncthreads();

  const int wave = tid >> 5;
  const int lane = tid & 31;
  const int h0 = lane >> 4, lr = lane & 15;
  const int mtiles = (M + 15) >> 4;
  const int mt = mblk * 8 + wave;
  if (mt >= mtiles) return;  // after barrier; no further syncs

  int row = mt * 16 + lr; if (row >= M) row = M - 1;  // branchless clamp
  const float* arow = A + (size_t)row * K;
  const v16h* bsw = (const v16h*)Bsw;

  v8f c0 = {}, c1 = {}, c2 = {}, c3 = {};
  for (int kc = 0; kc < K; kc += 32) {
    const v16h af = load_afrag_f32(arow, kc, h0);
    const int base = (kc >> 5) * 4 * 32 + lane;
    const v16h b0 = bsw[base + 0 * 32];
    const v16h b1 = bsw[base + 1 * 32];
    const v16h b2 = bsw[base + 2 * 32];
    const v16h b3 = bsw[base + 3 * 32];
    c0 = __builtin_amdgcn_wmma_f32_16x16x32_f16(false, af, false, b0, (short)0, c0, false, false);
    c1 = __builtin_amdgcn_wmma_f32_16x16x32_f16(false, af, false, b1, (short)0, c1, false, false);
    c2 = __builtin_amdgcn_wmma_f32_16x16x32_f16(false, af, false, b2, (short)0, c2, false, false);
    c3 = __builtin_amdgcn_wmma_f32_16x16x32_f16(false, af, false, b3, (short)0, c3, false, false);
  }

  v8f acc[4] = {c0, c1, c2, c3};
#pragma unroll
  for (int sub = 0; sub < 4; ++sub) {
    const int col = ng * 64 + sub * 16 + lr;
    const float bv = bias ? bias[col] : 0.0f;
#pragma unroll
    for (int i = 0; i < 8; ++i) {
      const int r = mt * 16 + i + 8 * h0;
      if (r < M) {
        float v = acc[sub][i] + bv;
        if (act) v = tanhf(v);
        if (rowscale) v *= rowscale[r];
        out[(size_t)r * Nc + col] = v;
      }
    }
  }
}

// ---------------------------------------------------------------------------
// Fused decompose + 3-way channel mix:
//   O[n][k][ij] = base?[n][k][ij] + sum_h I(T)[n][h][ij]*w0[h][k]
//                                 + A(T)[n][h][ij]*w1[h][k] + S(T)[n][h][ij]*w2[h][k]
// Block = 8 waves; the three 64x64 weights staged swizzled in LDS (f16, 24 KB).
// Each wave: 16 nodes x all 64 channels at fixed (i,j); I/A/S A-frags built
// once per K-chunk and reused across 4 column tiles (12 WMMAs per chunk).
// ---------------------------------------------------------------------------
__global__ void tn_mix(const float* __restrict__ T, const float* __restrict__ w0,
                       const float* __restrict__ w1, const float* __restrict__ w2,
                       const float* __restrict__ base, float* __restrict__ O) {
  __shared__ alignas(32) _Float16 Wsw[3 * HC * HC];  // 24 KB
  const int tid = threadIdx.x;
  {
    const float* wp[3] = {w0, w1, w2};
    for (int p = tid; p < 3 * HC * HC; p += 256) {
      const int wsel = p >> 12;       // /4096
      const int rem = p & 4095;
      const int k = rem >> 6, c = rem & 63;
      const int chunk = k >> 5, kr = k & 31;
      const int slot = (((wsel << 1) + chunk) * 4 + (c >> 4)) * 32 + ((kr >> 4) << 4) + (c & 15);
      Wsw[slot * 16 + (kr & 15)] = (_Float16)wp[wsel][rem];
    }
  }
  __syncthreads();

  const int NTN = N_NODES >> 4;  // 625
  const int wid = blockIdx.x * 8 + (tid >> 5);
  if (wid >= 9 * NTN) return;
  const int ij = wid / NTN;
  const int ntile = wid - ij * NTN;
  const int i = ij / 3, j = ij - 3 * i;
  const int lane = tid & 31;
  const int h0 = lane >> 4, lr = lane & 15;
  const int n = ntile * 16 + lr;
  const float* tn = T + (size_t)n * HC * 9;
  const v16h* wv = (const v16h*)Wsw;

  v8f acc[4] = {{}, {}, {}, {}};
#pragma unroll
  for (int chunk = 0; chunk < 2; ++chunk) {
    const int kc = chunk * 32;
    v16h aI, aA, aS;
#pragma unroll
    for (int idx = 0; idx < 16; ++idx) {
      const int h = kc + 16 * h0 + idx;
      const float* th = tn + (size_t)h * 9;
      const float tij = th[ij];
      const float tji = th[j * 3 + i];
      float Iv = 0.0f;
      if (i == j) Iv = (th[0] + th[4] + th[8]) * (1.0f / 3.0f);  // wave-uniform
      aI[idx] = (_Float16)Iv;
      aA[idx] = (_Float16)(0.5f * (tij - tji));
      aS[idx] = (_Float16)(0.5f * (tij + tji) - Iv);
    }
#pragma unroll
    for (int ct = 0; ct < 4; ++ct) {
      const v16h bI = wv[((0 * 2 + chunk) * 4 + ct) * 32 + lane];
      const v16h bA = wv[((1 * 2 + chunk) * 4 + ct) * 32 + lane];
      const v16h bS = wv[((2 * 2 + chunk) * 4 + ct) * 32 + lane];
      acc[ct] = __builtin_amdgcn_wmma_f32_16x16x32_f16(false, aI, false, bI, (short)0, acc[ct], false, false);
      acc[ct] = __builtin_amdgcn_wmma_f32_16x16x32_f16(false, aA, false, bA, (short)0, acc[ct], false, false);
      acc[ct] = __builtin_amdgcn_wmma_f32_16x16x32_f16(false, aS, false, bS, (short)0, acc[ct], false, false);
    }
  }
#pragma unroll
  for (int ct = 0; ct < 4; ++ct) {
    const int col = ct * 16 + lr;
#pragma unroll
    for (int q = 0; q < 8; ++q) {
      const int no = ntile * 16 + q + 8 * h0;
      const size_t o = ((size_t)no * HC + col) * 9 + ij;
      float v = acc[ct][q];
      if (base) v += base[o];
      O[o] = v;
    }
  }
}

// ---------------------------------------------------------------------------
// Elementwise / scatter kernels
// ---------------------------------------------------------------------------
__global__ void tn_cutoff(const float* __restrict__ w, float* __restrict__ C) {
  int e = blockIdx.x * blockDim.x + threadIdx.x;
  if (e >= N_EDGES) return;
  float x = w[e];
  C[e] = (x < CUT_R) ? 0.5f * (cosf(x * (PI_F / CUT_R)) + 1.0f) : 0.0f;
}

__global__ void tn_gatherZ(const int* __restrict__ z, const float* __restrict__ emb,
                           float* __restrict__ Z) {
  int idx = blockIdx.x * blockDim.x + threadIdx.x;
  if (idx >= N_NODES * HC) return;
  int n = idx / HC, h = idx - n * HC;
  Z[idx] = emb[(size_t)z[n] * HC + h];
}

__global__ void tn_zcat(const int* __restrict__ ei, const float* __restrict__ Z,
                        float* __restrict__ Zc) {
  int idx = blockIdx.x * blockDim.x + threadIdx.x;
  if (idx >= N_EDGES * 2 * HC) return;
  int e = idx >> 7, c = idx & 127;
  int node = (c < HC) ? ei[N_EDGES + e] : ei[e];  // [Z[dst], Z[src]]
  Zc[idx] = Z[(size_t)node * HC + (c & (HC - 1))];
}

__global__ void tn_scatter_emb(const int* __restrict__ ei, const float* __restrict__ ew,
                               const float* __restrict__ ev, const float* __restrict__ W1,
                               const float* __restrict__ W2, const float* __restrict__ Zij,
                               float* __restrict__ X) {
  int idx = blockIdx.x * blockDim.x + threadIdx.x;
  if (idx >= N_EDGES * HC) return;
  int e = idx / HC;
  int d = ei[N_EDGES + e];
  float w = ew[e];
  float ux = ev[e * 3] / w, uy = ev[e * 3 + 1] / w, uz = ev[e * 3 + 2] / w;
  float c1 = W1[idx], c2 = W2[idx], zc = Zij[idx];
  float u[3] = {ux, uy, uz};
  float sk[9] = {0.f, -uz, uy, uz, 0.f, -ux, -uy, ux, 0.f};
  float* dst = X + ((size_t)d * HC + (idx - e * HC)) * 9;
#pragma unroll
  for (int i = 0; i < 3; ++i)
#pragma unroll
    for (int j = 0; j < 3; ++j)
      unsafeAtomicAdd(&dst[i * 3 + j], zc * (c1 * sk[i * 3 + j] + c2 * u[i] * u[j]));
}

__global__ void tn_norms(const float* __restrict__ T, float* __restrict__ nrm) {
  int idx = blockIdx.x * blockDim.x + threadIdx.x;
  if (idx >= N_NODES * HC) return;
  int n = idx / HC, h = idx - n * HC;
  const float* t = T + (size_t)idx * 9;
  float t0=t[0],t1=t[1],t2=t[2],t3=t[3],t4=t[4],t5=t[5],t6=t[6],t7=t[7],t8=t[8];
  float dm = (t0 + t4 + t8) * (1.0f / 3.0f);
  float In = 3.0f * dm * dm;
  float a01 = 0.5f * (t1 - t3), a02 = 0.5f * (t2 - t6), a12 = 0.5f * (t5 - t7);
  float An = 2.0f * (a01 * a01 + a02 * a02 + a12 * a12);
  float s00 = t0 - dm, s11 = t4 - dm, s22 = t8 - dm;
  float s01 = 0.5f * (t1 + t3), s02 = 0.5f * (t2 + t6), s12 = 0.5f * (t5 + t7);
  float Sn = s00*s00 + s11*s11 + s22*s22 + 2.0f * (s01*s01 + s02*s02 + s12*s12);
  nrm[(size_t)n * H3 + h] = In;
  nrm[(size_t)n * H3 + HC + h] = An;
  nrm[(size_t)n * H3 + 2 * HC + h] = Sn;
}

__device__ __forceinline__ void nrt9(const float* t, float f0, float f1, float f2, float* o) {
  float dm = (t[0] + t[4] + t[8]) * (1.0f / 3.0f);
  float d = (f0 - f2) * dm;
  float p = 0.5f * (f1 + f2), m = 0.5f * (f1 - f2);
#pragma unroll
  for (int i = 0; i < 3; ++i)
#pragma unroll
    for (int j = 0; j < 3; ++j)
      o[i * 3 + j] = ((i == j) ? d : 0.0f) + p * t[i * 3 + j] - m * t[j * 3 + i];
}

// mode 0: X = nrt(T, f).  mode 1: dX = nrt(T, f); X += dX - 0.5*dX@dX
__global__ void tn_nrt(const float* __restrict__ T, const float* __restrict__ f,
                       float* __restrict__ X, int mode) {
  int idx = blockIdx.x * blockDim.x + threadIdx.x;
  if (idx >= N_NODES * HC) return;
  int n = idx / HC, h = idx - n * HC;
  float t[9], o[9];
#pragma unroll
  for (int k = 0; k < 9; ++k) t[k] = T[(size_t)idx * 9 + k];
  const float* fp = f + (size_t)n * H3 + h * 3;
  nrt9(t, fp[0], fp[1], fp[2], o);
  if (mode == 0) {
#pragma unroll
    for (int k = 0; k < 9; ++k) X[(size_t)idx * 9 + k] = o[k];
  } else {
    float q[9];
#pragma unroll
    for (int i = 0; i < 3; ++i)
#pragma unroll
      for (int j = 0; j < 3; ++j)
        q[i*3+j] = o[i*3]*o[j] + o[i*3+1]*o[3+j] + o[i*3+2]*o[6+j];
#pragma unroll
    for (int k = 0; k < 9; ++k) X[(size_t)idx * 9 + k] += o[k] - 0.5f * q[k];
  }
}

// M[dst] += nrt(Y[src], ea)   (segment_sum via f32 atomics; Y is L2-resident)
__global__ void tn_message(const int* __restrict__ ei, const float* __restrict__ Y,
                           const float* __restrict__ ea, float* __restrict__ Mb) {
  int idx = blockIdx.x * blockDim.x + threadIdx.x;
  if (idx >= N_EDGES * HC) return;
  int e = idx / HC, h = idx - e * HC;
  int s = ei[e], d = ei[N_EDGES + e];
  const float* ty = Y + ((size_t)s * HC + h) * 9;
  float t[9], o[9];
#pragma unroll
  for (int k = 0; k < 9; ++k) t[k] = ty[k];
  const float* fp = ea + (size_t)e * H3 + h * 3;
  nrt9(t, fp[0], fp[1], fp[2], o);
  float* dst = Mb + ((size_t)d * HC + h) * 9;
#pragma unroll
  for (int k = 0; k < 9; ++k) unsafeAtomicAdd(&dst[k], o[k]);
}

// Y = M @ (Y @ M), per (n,h) 3x3
__global__ void tn_ymm(float* __restrict__ Y, const float* __restrict__ Mb) {
  int idx = blockIdx.x * blockDim.x + threadIdx.x;
  if (idx >= N_NODES * HC) return;
  float y[9], m[9], t[9];
#pragma unroll
  for (int k = 0; k < 9; ++k) { y[k] = Y[(size_t)idx*9+k]; m[k] = Mb[(size_t)idx*9+k]; }
#pragma unroll
  for (int i = 0; i < 3; ++i)
#pragma unroll
    for (int j = 0; j < 3; ++j)
      t[i*3+j] = y[i*3]*m[j] + y[i*3+1]*m[3+j] + y[i*3+2]*m[6+j];
#pragma unroll
  for (int i = 0; i < 3; ++i)
#pragma unroll
    for (int j = 0; j < 3; ++j)
      Y[(size_t)idx*9 + i*3+j] = m[i*3]*t[j] + m[i*3+1]*t[3+j] + m[i*3+2]*t[6+j];
}

// O = T / (tnorm(T) + 1)
__global__ void tn_tnormdiv(const float* __restrict__ T, float* __restrict__ O) {
  int idx = blockIdx.x * blockDim.x + threadIdx.x;
  if (idx >= N_NODES * HC) return;
  float t[9], ss = 0.0f;
#pragma unroll
  for (int k = 0; k < 9; ++k) { t[k] = T[(size_t)idx*9+k]; ss += t[k]*t[k]; }
  float inv = 1.0f / (ss + 1.0f);
#pragma unroll
  for (int k = 0; k < 9; ++k) O[(size_t)idx*9+k] = t[k] * inv;
}

// ---------------------------------------------------------------------------
// Host side
// ---------------------------------------------------------------------------
static inline int g1(long n) { return (int)((n + 255) / 256); }

static void launch_gemm(const float* A, const float* W, const float* b, const float* rs,
                        float* out, int M, int K, int Nc, int act, hipStream_t s) {
  const int ngroups = Nc / 64;
  const int mblks = ((M + 15) / 16 + 7) / 8;
  tn_gemm<<<mblks * ngroups, 256, 0, s>>>(A, W, b, rs, out, M, K, Nc, act, ngroups);
}
static void launch_mix(const float* T, const float* w0, const float* w1, const float* w2,
                       const float* base, float* O, hipStream_t s) {
  const int waves = 9 * (N_NODES / 16);
  tn_mix<<<(waves + 7) / 8, 256, 0, s>>>(T, w0, w1, w2, base, O);
}

extern "C" void kernel_launch(void* const* d_in, const int* in_sizes, int n_in,
                              void* d_out, int out_size, void* d_ws, size_t ws_size,
                              hipStream_t stream) {
  if (n_in < 72) return;  // expects recursive insertion-order flatten of setup_inputs()
  auto F = [&](int i) { return (const float*)d_in[i]; };
  const int* z  = (const int*)d_in[0];
  const int* ei = (const int*)d_in[1];
  const float* ew = F(2); const float* ev = F(3); const float* attr = F(4);
  const float* emb = F(5);
  const float* dp1w = F(6), *dp1b = F(7), *dp2w = F(8), *dp2b = F(9);
  const float* e2w = F(10), *e2b = F(11);
  const float* elt0 = F(12), *elt1 = F(13), *elt2 = F(14);
  const float* es0w = F(15), *es0b = F(16), *es1w = F(17), *es1b = F(18);
  const float* outw = F(70), *outb = F(71);

  float* ws = (float*)d_ws;
  const size_t NH9 = (size_t)N_NODES * HC * 9;
  float* Zb = ws;
  float* Xb = Zb + (size_t)N_NODES * HC;
  float* Yb = Xb + NH9;
  float* Mb = Yb + NH9;
  float* n1 = Mb + NH9;
  float* n2 = n1 + (size_t)N_NODES * H3;
  float* Cb = n2 + (size_t)N_NODES * H3;
  float* E1 = Cb + N_EDGES;                 // E x 192
  float* E2 = E1 + (size_t)N_EDGES * H3;    // E x 192

  const int TB = 256;
  const long NHn = (long)N_NODES * HC;
  const long EHn = (long)N_EDGES * HC;

  // ---- embedding block ----
  tn_cutoff<<<g1(N_EDGES), TB, 0, stream>>>(ew, Cb);
  tn_gatherZ<<<g1(NHn), TB, 0, stream>>>(z, emb, Zb);
  hipMemsetAsync(Xb, 0, NH9 * sizeof(float), stream);
  launch_gemm(attr, dp1w, dp1b, Cb, E1, N_EDGES, RB, HC, 0, stream);                       // W1
  launch_gemm(attr, dp2w, dp2b, Cb, E1 + (size_t)N_EDGES * HC, N_EDGES, RB, HC, 0, stream);// W2
  tn_zcat<<<g1((long)N_EDGES * 2 * HC), TB, 0, stream>>>(ei, Zb, E2);
  launch_gemm(E2, e2w, e2b, nullptr, E1 + 2 * (size_t)N_EDGES * HC, N_EDGES, 2 * HC, HC, 0, stream);
  tn_scatter_emb<<<g1(EHn), TB, 0, stream>>>(ei, ew, ev, E1, E1 + (size_t)N_EDGES * HC,
                                             E1 + 2 * (size_t)N_EDGES * HC, Xb);
  tn_norms<<<g1(NHn), TB, 0, stream>>>(Xb, n1);
  launch_mix(Xb, elt0, elt1, elt2, nullptr, Mb, stream);
  launch_gemm(n1, es0w, es0b, nullptr, n2, N_NODES, H3, H3, 1, stream);
  launch_gemm(n2, es1w, es1b, nullptr, n1, N_NODES, H3, H3, 1, stream);
  tn_nrt<<<g1(NHn), TB, 0, stream>>>(Mb, n1, Xb, 0);

  // ---- message-passing layers ----
  for (int l = 0; l < 3; ++l) {
    const int B = 19 + 17 * l;
    const float* p0w = F(B+0), *p0b = F(B+1), *p1w = F(B+2), *p1b = F(B+3);
    const float* WI1 = F(B+4), *WA1 = F(B+5), *WS1 = F(B+6);
    const float* WI2 = F(B+7), *WA2 = F(B+8), *WS2 = F(B+9);
    const float* l0 = F(B+10), *l1 = F(B+11), *l2 = F(B+12);
    const float* s0w = F(B+13), *s0b = F(B+14), *s1w = F(B+15), *s1b = F(B+16);

    launch_gemm(attr, p0w, p0b, nullptr, E1, N_EDGES, RB, H3, 1, stream);   // tanh(attr@W0+b)
    launch_gemm(E1, p1w, p1b, Cb, E2, N_EDGES, H3, H3, 1, stream);          // tanh(.)@W1 * C
    launch_mix(Xb, WI1, WA1, WS1, nullptr, Yb, stream);                     // Y = mixes of X
    hipMemsetAsync(Mb, 0, NH9 * sizeof(float), stream);
    tn_message<<<g1(EHn), TB, 0, stream>>>(ei, Yb, E2, Mb);                 // M = segsum(nrt)
    tn_ymm<<<g1(NHn), TB, 0, stream>>>(Yb, Mb);                             // Y = M@(Y@M)
    launch_mix(Yb, WI2, WA2, WS2, Xb, Mb, stream);                          // tmp = X + mixes
    tn_tnormdiv<<<g1(NHn), TB, 0, stream>>>(Mb, Yb);                        // Y = tmp/(tnorm+1)
    tn_norms<<<g1(NHn), TB, 0, stream>>>(Yb, n1);
    launch_mix(Yb, l0, l1, l2, nullptr, Mb, stream);
    launch_gemm(n1, s0w, s0b, nullptr, n2, N_NODES, H3, H3, 1, stream);
    launch_gemm(n2, s1w, s1b, nullptr, n1, N_NODES, H3, H3, 1, stream);
    tn_nrt<<<g1(NHn), TB, 0, stream>>>(Mb, n1, Xb, 1);                      // X += dX-0.5dX^2
  }

  // ---- output head ----
  tn_norms<<<g1(NHn), TB, 0, stream>>>(Xb, n1);
  launch_gemm(n1, outw, outb, nullptr, (float*)d_out, N_NODES, H3, HC, 1, stream);
}
